// MultiHeadedAttention_19043884990727
// MI455X (gfx1250) — matrix operand order repaired and checked by reading, not compiled, not run on hardware
//
#include <hip/hip_runtime.h>
#include <hip/hip_bf16.h>

// ---------------------------------------------------------------------------
// MultiHeadedAttention for gfx1250 (MI455X): bf16 WMMA + TDM tile staging,
// software-pipelined (double-buffered) LDS stages.
// B=8, S=1024, D=512, H=8, DK=64.
// Pipeline:
//   1) gemm_bias<false,true>  : Q/K/V projections fp32->bf16, head-major out
//   2) attn_kernel            : flash-attention; K tiles staged via
//                               tensor_load_to_lds (TENSORcnt) double-buffered
//                               against WMMA compute; online softmax
//   3) gemm_bias<true,false>  : output projection bf16 ctx -> fp32 d_out
// ---------------------------------------------------------------------------

typedef __attribute__((ext_vector_type(16))) __bf16 v16bf;
typedef __attribute__((ext_vector_type(8)))  float  v8f;
typedef __attribute__((ext_vector_type(4)))  unsigned int u32x4;
typedef __attribute__((ext_vector_type(4)))  int    i32x4;
typedef __attribute__((ext_vector_type(8)))  int    i32x8;

#if defined(__gfx1250__) && __has_builtin(__builtin_amdgcn_tensor_load_to_lds)
#define USE_TDM 1
#pragma message("CDNA5: TDM path ENABLED (tensor_load_to_lds)")
#else
#define USE_TDM 0
#pragma message("CDNA5: TDM path DISABLED (cooperative-load fallback)")
#endif

namespace {
constexpr int Bc  = 8;
constexpr int Sc  = 1024;
constexpr int Dc  = 512;
constexpr int Hc  = 8;
constexpr int DKc = 64;

constexpr int TM = 128, TN = 128, TK = 32;   // GEMM tile
constexpr int BQ = 128, BK = 64;             // attention tiles
constexpr int PAD = 8;                       // LDS minor-dim pad (bank spread)
constexpr int NB = Sc / BK;                  // key blocks per head
}

__device__ __forceinline__ v8f wmma_bf16(v16bf a, v16bf b, v8f c) {
  return __builtin_amdgcn_wmma_f32_16x16x32_bf16(
      /*neg_a=*/false, a, /*neg_b=*/false, b,
      /*c_mod=*/(short)0, c, /*reuse_a=*/false, /*reuse_b=*/false);
}

// A-fragment (16x32, bf16) from row-major LDS tile: rows 0..15 at `base`,
// columns kOff..kOff+31, leading dim ld (elements).
// ISA layout: lane r (half g): element e -> K = (e&8)*2 + (e&7) + 8*g.
__device__ __forceinline__ v16bf load_a_frag(const __bf16* base, int ld,
                                             int kOff, int r, int g) {
  v16bf a;
#pragma unroll
  for (int e = 0; e < 16; ++e) {
    int k = (e & 8) * 2 + (e & 7) + 8 * g;
    a[e] = base[r * ld + kOff + k];
  }
  return a;
}

// B-fragment (32x16, bf16) where LDS holds B^T row-major: Bt[n][k].
// element e -> K = (e&8)*2 + (e&7) + 8*g, column N = r (matches A layout).
__device__ __forceinline__ v16bf load_bt_frag(const __bf16* baseT, int ld,
                                              int kOff, int r, int g) {
  v16bf b;
#pragma unroll
  for (int e = 0; e < 16; ++e) {
    b[e] = baseT[r * ld + kOff + (e & 8) * 2 + (e & 7) + 8 * g];
  }
  return b;
}

#if USE_TDM
// Issue one 2D TDM tile load: `rows` x `rowlen` bf16 elements from a
// row-major tensor (leading dim `row_stride` elements) into LDS at byte
// offset `lds_off`, inserting PAD bf16 (16 B = 4 DWORDs) after every
// 128-byte row (rowlen must be 64 elements). Tracked on TENSORcnt.
__device__ __forceinline__ void tdm_load_2d(const void* gptr, unsigned lds_off,
                                            int rows, int rowlen,
                                            int row_stride, int tensor_rows) {
  unsigned long long ga = (unsigned long long)gptr;
  u32x4 g0;
  g0[0] = 1u;                                              // count=1 (valid)
  g0[1] = lds_off;                                         // lds_addr
  g0[2] = (unsigned)(ga & 0xffffffffu);                    // global_addr lo
  g0[3] = (unsigned)((ga >> 32) & 0x01ffffffu) | (2u << 30);  // hi | type=2
  i32x8 g1;
  // data_size=1 (2B) | pad_enable | pad_interval=4 (32 DW) | pad_amount=3 (4 DW)
  g1[0] = (1 << 16) | (1 << 20) | (4 << 22) | (3 << 25);
  g1[1] = (rowlen & 0xffff) << 16;        // tensor_dim0[15:0] @ bits 79:48
  g1[2] = (tensor_rows & 0xffff) << 16;   // tensor_dim1[15:0] @ bits 111:80
  g1[3] = (rowlen & 0xffff) << 16;        // tile_dim0 @ bits 127:112
  g1[4] = rows & 0xffff;                  // tile_dim1 (tile_dim2 = 0)
  g1[5] = row_stride;                     // tensor_dim0_stride[31:0]
  g1[6] = 0;                              // stride hi / dim1_stride lo
  g1[7] = 0;                              // dim1_stride hi
  i32x4 z4 = {0, 0, 0, 0};
#if __clang_major__ >= 23
  i32x8 z8 = {0, 0, 0, 0, 0, 0, 0, 0};
  __builtin_amdgcn_tensor_load_to_lds(g0, g1, z4, z4, z8, 0);
#else
  __builtin_amdgcn_tensor_load_to_lds(g0, g1, z4, z4, 0);
#endif
}
#endif  // USE_TDM

// ---------------------------------------------------------------------------
// GEMM:  Y = X @ W + bias.  X: M x K (fp32 or bf16, row-major),
// W: K x N fp32 row-major (reference stores weights as (in,out)).
// HEADMAJOR=true  -> write bf16 to (B,H,S,DK) layout (for QKV projections).
// HEADMAJOR=false -> write fp32 row-major (final output).
// Workgroup: 256 threads = 8 waves; tile 128x128; wave w owns rows 16w..16w+15.
// Double-buffered K-tiles: global loads of tile kt+1 overlap WMMA of tile kt.
// ---------------------------------------------------------------------------
template <bool IN_BF16, bool HEADMAJOR>
__global__ __launch_bounds__(256) void gemm_bias_kernel(
    const void* __restrict__ Xv, const float* __restrict__ W,
    const float* __restrict__ bias, void* __restrict__ Yv,
    int Mdim, int Ndim, int Kdim) {
  __shared__ __bf16 sA[2][TM][TK + PAD];   // X tile, row-major
  __shared__ __bf16 sBt[2][TN][TK + PAD];  // W tile, transposed: [n][k]

  const int tid  = threadIdx.x;
  const int w    = tid >> 5;
  const int lane = tid & 31;
  const int g    = lane >> 4;
  const int r    = lane & 15;
  const int tm   = blockIdx.x * TM;
  const int tn   = blockIdx.y * TN;

  const float*  Xf = (const float*)Xv;
  const __bf16* Xb = (const __bf16*)Xv;

  auto loadTile = [&](int kt, int buf) {
    for (int idx = tid; idx < TM * TK; idx += 256) {
      int row = idx >> 5, col = idx & 31;
      float v = IN_BF16 ? (float)Xb[(size_t)(tm + row) * Kdim + kt + col]
                        : Xf[(size_t)(tm + row) * Kdim + kt + col];
      sA[buf][row][col] = (__bf16)v;
    }
    for (int idx = tid; idx < TK * TN; idx += 256) {
      int nn = idx & (TN - 1), kk = idx >> 7;
      sBt[buf][nn][kk] = (__bf16)W[(size_t)(kt + kk) * Ndim + tn + nn];
    }
  };

  v8f acc[8];
#pragma unroll
  for (int i = 0; i < 8; ++i)
#pragma unroll
    for (int e = 0; e < 8; ++e) acc[i][e] = 0.f;

  loadTile(0, 0);
  __syncthreads();

  for (int kt = 0; kt < Kdim; kt += TK) {
    const int cur = (kt / TK) & 1;
    const bool more = (kt + TK) < Kdim;
    if (more) loadTile(kt + TK, cur ^ 1);  // overlap with WMMA below

    v16bf a = load_a_frag(&sA[cur][16 * w][0], TK + PAD, 0, r, g);
#pragma unroll
    for (int ni = 0; ni < 8; ++ni) {
      v16bf b = load_bt_frag(&sBt[cur][16 * ni][0], TK + PAD, 0, r, g);
      acc[ni] = wmma_bf16(a, b, acc[ni]);
    }
    if (more) __syncthreads();  // publish tile kt+TK; cur buf free next iter
  }

  // Epilogue: C element j -> row 16w + j + 8g, col 16ni + r.
#pragma unroll
  for (int ni = 0; ni < 8; ++ni) {
#pragma unroll
    for (int j = 0; j < 8; ++j) {
      int row = tm + 16 * w + j + 8 * g;
      int col = tn + 16 * ni + r;
      float v = acc[ni][j] + bias[col];
      if (HEADMAJOR) {
        int bb = row >> 10;          // / Sc
        int ss = row & (Sc - 1);
        int hh = col >> 6;           // / DKc
        int dd = col & (DKc - 1);
        ((__bf16*)Yv)[((((size_t)bb * Hc + hh) * Sc) + ss) * DKc + dd] =
            (__bf16)v;
      } else {
        ((float*)Yv)[(size_t)row * Ndim + col] = v;
      }
    }
  }
}

// ---------------------------------------------------------------------------
// Flash attention. Grid: (B*H, S/BQ). Block: 256 threads (8 waves).
// Wave w owns query rows 16w..16w+15 of the 128-query tile.
// K tiles staged by the Tensor Data Mover (wave 0 issues, TENSORcnt waited
// before the publishing barrier); V cooperatively transposed into LDS.
// Two-stage pipeline: stage kb+1 (TDM + V loads) overlaps compute of kb.
// ---------------------------------------------------------------------------
struct AttnSmem {
  __bf16 sQ[BQ][DKc + PAD];       // queries, row-major (TDM pads rows)
  __bf16 sK[2][BK][DKc + PAD];    // keys, row-major (= B^T for QK^T)
  __bf16 sVt[2][DKc][BK + PAD];   // V transposed (= B^T for PV)
  __bf16 sP[8][16][BK + PAD];     // wave-private probability tiles
};

__global__ __launch_bounds__(256) void attn_kernel(
    const __bf16* __restrict__ Qh, const __bf16* __restrict__ Kh,
    const __bf16* __restrict__ Vh, const int* __restrict__ mask,
    __bf16* __restrict__ ctx) {
  __shared__ AttnSmem sm;

  const int tid  = threadIdx.x;
  const int w    = tid >> 5;
  const int lane = tid & 31;
  const int g    = lane >> 4;
  const int r    = lane & 15;
  const int bh   = blockIdx.x;
  const int qt   = blockIdx.y;
  const int bb   = bh / Hc;
  const int hh   = bh % Hc;
  const size_t base = (size_t)bh * Sc * DKc;

  auto issueK = [&](int kb, int buf) {
#if USE_TDM
    if (w == 0) {
      unsigned lds = (unsigned)(unsigned long long)(void*)&sm.sK[buf][0][0];
      tdm_load_2d(Kh + base + (size_t)(kb * BK) * DKc, lds, BK, DKc, DKc, Sc);
    }
#else
    for (int idx = tid; idx < BK * DKc; idx += 256) {
      int row = idx >> 6, col = idx & 63;
      sm.sK[buf][row][col] = Kh[base + (size_t)(kb * BK + row) * DKc + col];
    }
#endif
  };
  auto issueV = [&](int kb, int buf) {
    for (int idx = tid; idx < BK * DKc; idx += 256) {
      int row = idx >> 6, col = idx & 63;  // row=key, col=feature
      sm.sVt[buf][col][row] = Vh[base + (size_t)(kb * BK + row) * DKc + col];
    }
  };

  // --- prologue: Q tile + stage 0 ---
#if USE_TDM
  if (w == 0) {
    unsigned ldsQ = (unsigned)(unsigned long long)(void*)&sm.sQ[0][0];
    tdm_load_2d(Qh + base + (size_t)(qt * BQ) * DKc, ldsQ, BQ, DKc, DKc, Sc);
  }
#else
  for (int idx = tid; idx < BQ * DKc; idx += 256) {
    int row = idx >> 6, col = idx & 63;
    sm.sQ[row][col] = Qh[base + (size_t)(qt * BQ + row) * DKc + col];
  }
#endif
  issueK(0, 0);
  issueV(0, 0);
#if USE_TDM
  if (w == 0) __builtin_amdgcn_s_wait_tensorcnt(0);
#endif
  __syncthreads();  // publish Q + stage 0

  v8f accO[4];
  float rowmax[8], rowsum[8];
#pragma unroll
  for (int no = 0; no < 4; ++no)
#pragma unroll
    for (int e = 0; e < 8; ++e) accO[no][e] = 0.f;
#pragma unroll
  for (int j = 0; j < 8; ++j) { rowmax[j] = -1e30f; rowsum[j] = 0.f; }

  for (int kb = 0; kb < NB; ++kb) {
    const int cur = kb & 1;
    const bool more = (kb + 1) < NB;

    // Stage kb+1 in flight while computing stage kb.
    if (more) {
      issueK(kb + 1, cur ^ 1);
      issueV(kb + 1, cur ^ 1);
      if (kb + 2 < NB) {
        const char* nv =
            (const char*)(Vh + base + (size_t)((kb + 2) * BK) * DKc);
        __builtin_prefetch(nv + tid * 32, 0, 3);
      }
    }

    // --- scores: S = Q (16xDK) @ K^T -> 16 x 64 (4 subtiles) ---
    v8f sfr[4];
#pragma unroll
    for (int ni = 0; ni < 4; ++ni)
#pragma unroll
      for (int e = 0; e < 8; ++e) sfr[ni][e] = 0.f;

#pragma unroll
    for (int kc = 0; kc < DKc; kc += 32) {
      v16bf a = load_a_frag(&sm.sQ[16 * w][0], DKc + PAD, kc, r, g);
#pragma unroll
      for (int ni = 0; ni < 4; ++ni) {
        v16bf b = load_bt_frag(&sm.sK[cur][16 * ni][0], DKc + PAD, kc, r, g);
        sfr[ni] = wmma_bf16(a, b, sfr[ni]);
      }
    }

    // Hoisted mask loads: key index is independent of j.
    int keep[4];
#pragma unroll
    for (int ni = 0; ni < 4; ++ni)
      keep[ni] = mask[bb * Sc + kb * BK + 16 * ni + r];

    // --- online softmax (per query row j+8g; lanes r=0..15 share a row) ---
#pragma unroll
    for (int j = 0; j < 8; ++j) {
      float vals[4];
#pragma unroll
      for (int ni = 0; ni < 4; ++ni)
        vals[ni] = sfr[ni][j] * 0.125f;  // 1/sqrt(DK)

      float mloc = -1e30f;
#pragma unroll
      for (int ni = 0; ni < 4; ++ni)
        mloc = fmaxf(mloc, keep[ni] ? vals[ni] : -1e30f);
#pragma unroll
      for (int off = 1; off < 16; off <<= 1)
        mloc = fmaxf(mloc, __shfl_xor(mloc, off, 16));

      float mnew = fmaxf(rowmax[j], mloc);
      float corr = __expf(rowmax[j] - mnew);
      rowmax[j] = mnew;

      float ps = 0.f;
#pragma unroll
      for (int ni = 0; ni < 4; ++ni) {
        float p = keep[ni] ? __expf(vals[ni] - mnew) : 0.f;
        ps += p;
        sm.sP[w][j + 8 * g][16 * ni + r] = (__bf16)p;  // wave-private
      }
#pragma unroll
      for (int off = 1; off < 16; off <<= 1) ps += __shfl_xor(ps, off, 16);
      rowsum[j] = rowsum[j] * corr + ps;
#pragma unroll
      for (int no = 0; no < 4; ++no) accO[no][j] *= corr;
    }

    // --- accumulate O += P (16x64) @ V (64x64) ---
#pragma unroll
    for (int kc = 0; kc < BK; kc += 32) {
      v16bf a = load_a_frag(&sm.sP[w][0][0], BK + PAD, kc, r, g);
#pragma unroll
      for (int no = 0; no < 4; ++no) {
        v16bf b = load_bt_frag(&sm.sVt[cur][16 * no][0], BK + PAD, kc, r, g);
        accO[no] = wmma_bf16(a, b, accO[no]);
      }
    }

    if (more) {
#if USE_TDM
      if (w == 0) __builtin_amdgcn_s_wait_tensorcnt(0);
#endif
      __syncthreads();  // publish stage kb+1; cur buffers free next iter
    }
  }

  // Epilogue: normalize and write context in (B,S,D) layout (bf16).
#pragma unroll
  for (int j = 0; j < 8; ++j) {
    float inv = 1.f / fmaxf(rowsum[j], 1e-20f);
    int qrow = qt * BQ + 16 * w + j + 8 * g;
#pragma unroll
    for (int no = 0; no < 4; ++no) {
      int f = 16 * no + r;
      float v = accO[no][j] * inv;
      ctx[((size_t)bb * Sc + qrow) * Dc + hh * DKc + f] = (__bf16)v;
    }
  }
}

// ---------------------------------------------------------------------------
extern "C" void kernel_launch(void* const* d_in, const int* in_sizes, int n_in,
                              void* d_out, int out_size, void* d_ws,
                              size_t ws_size, hipStream_t stream) {
  const float* query = (const float*)d_in[0];
  const float* key   = (const float*)d_in[1];
  const float* value = (const float*)d_in[2];
  const int*   mask  = (const int*)d_in[3];
  const float* Wq = (const float*)d_in[4];
  const float* bq = (const float*)d_in[5];
  const float* Wk = (const float*)d_in[6];
  const float* bk = (const float*)d_in[7];
  const float* Wv = (const float*)d_in[8];
  const float* bv = (const float*)d_in[9];
  const float* Wo = (const float*)d_in[10];
  const float* bo = (const float*)d_in[11];
  float* out = (float*)d_out;

  // Workspace: qh | kh | vh (head-major bf16) | ctx (row-major bf16) = 32 MB.
  const size_t NE = (size_t)Bc * Hc * Sc * DKc;  // 4M elements
  __bf16* qh  = (__bf16*)d_ws;
  __bf16* kh  = qh + NE;
  __bf16* vh  = kh + NE;
  __bf16* ctx = vh + NE;

  const int M = Bc * Sc;  // 8192
  dim3 blk(256);
  dim3 gGemm(M / TM, Dc / TN);  // (64, 4)

  gemm_bias_kernel<false, true><<<gGemm, blk, 0, stream>>>(
      query, Wq, bq, qh, M, Dc, Dc);
  gemm_bias_kernel<false, true><<<gGemm, blk, 0, stream>>>(
      key, Wk, bk, kh, M, Dc, Dc);
  gemm_bias_kernel<false, true><<<gGemm, blk, 0, stream>>>(
      value, Wv, bv, vh, M, Dc, Dc);

  dim3 gAttn(Bc * Hc, Sc / BQ);  // (64, 8)
  attn_kernel<<<gAttn, blk, 0, stream>>>(qh, kh, vh, mask, ctx);

  gemm_bias_kernel<true, false><<<gGemm, blk, 0, stream>>>(
      ctx, Wo, bo, out, M, Dc, Dc);
}
